// MoE_42356967473647
// MI455X (gfx1250) — matrix operand order repaired and checked by reading, not compiled, not run on hardware
//
#include <hip/hip_runtime.h>
#include <hip/hip_bf16.h>
#include <math.h>

// ---------------------------------------------------------------------------
// MoE forward for MI455X (gfx1250, wave32, WMMA).
// B=4096, D=1024, H=4096, O=1024, E=8, K=4
// 550 GFLOP of expert GEMMs on v_wmma_f32_16x16x32_bf16.
// Weights pre-swizzled once per launch into WMMA B-fragment layout (bf16):
// the GEMM hot loop does contiguous global_load_b128 only. Gates are folded
// into h so phase 2 accumulates y directly (no per-expert oacc).
// M=32 tokens per workgroup; expert dim split 2-ways for occupancy.
// ---------------------------------------------------------------------------

#define BB 4096
#define DD 1024
#define HH 4096
#define OO 1024
#define EE 8
#define KK 4
#define MT 32           // tokens per workgroup
#define HC 256          // H-chunk staged in LDS per phase
#define NOISE_EPS 0.01f
#define LOSS_COEF 0.01f

#define KT1 (DD / 32)   // 32  k-tiles in W1
#define NT1 (HH / 16)   // 256 n-tiles in W1
#define KT2 (HH / 32)   // 128 k-tiles in W2
#define NT2 (OO / 16)   // 64  n-tiles in W2
#define TILE_ELEMS 512  // 32x16 bf16 per fragment tile (1 KB)

typedef __attribute__((ext_vector_type(16))) __bf16 bf16x16;
typedef __attribute__((ext_vector_type(8)))  float  f32x8;

// ---- WMMA helpers ----------------------------------------------------------

__device__ __forceinline__ f32x8 wmma_bf16(bf16x16 a, bf16x16 b, f32x8 c) {
    return __builtin_amdgcn_wmma_f32_16x16x32_bf16(
        false, a, false, b, (short)0, c, false, false);
}

// A-matrix 16x32 bf16 fragment from row-major LDS [16][ld].
__device__ __forceinline__ bf16x16 load_a_lds(const __bf16* base, int ld,
                                              int k0, int lane) {
    const int row = lane & 15;
    const int hb  = (lane < 16) ? 0 : 16;
    const __bf16* p = base + row * ld + k0 + hb;
    union { bf16x16 v; uint4 q[2]; } u;
    u.q[0] = *(const uint4*)(p);
    u.q[1] = *(const uint4*)(p + 8);
    return u.v;
}

// B fragment from pre-swizzled bf16 tile: lane reads 32 contiguous bytes.
__device__ __forceinline__ bf16x16 load_b_swz(const __bf16* __restrict__ t,
                                              int lane) {
    union { bf16x16 v; uint4 q[2]; } u;
    const uint4* p = (const uint4*)t + lane * 2;
    u.q[0] = p[0];
    u.q[1] = p[1];
    return u.v;
}

// Fallback: B fragment streamed from global f32 weights.
__device__ __forceinline__ bf16x16 load_b_f32(const float* __restrict__ W,
                                              int ld, int lane) {
    const int col = lane & 15;
    const int kb  = (lane < 16) ? 0 : 16;
    const float* p = W + (size_t)kb * ld + col;
    bf16x16 b;
#pragma unroll
    for (int j = 0; j < 16; ++j) b[j] = (__bf16)p[(size_t)j * ld];
    return b;
}

// ---- weight pre-swizzle: f32 [E][K][N] -> bf16 fragment tiles --------------

__global__ __launch_bounds__(256)
void moe_swz_kernel(const float* __restrict__ src, __bf16* __restrict__ dst,
                    int KT, int NT, int ld, int ntiles_total) {
    const int lane = threadIdx.x & 31;
    const int tile = blockIdx.x * 8 + (threadIdx.x >> 5);
    if (tile >= ntiles_total) return;
    const int nt = tile % NT;
    const int kt = (tile / NT) % KT;
    const int e  = tile / (NT * KT);
    const int col = lane & 15;
    const int kb  = (lane < 16) ? 0 : 16;
    const float* p = src + (size_t)e * KT * 32 * ld
                         + (size_t)(kt * 32 + kb) * ld + nt * 16 + col;
    union { bf16x16 v; uint4 q[2]; } u;
#pragma unroll
    for (int j = 0; j < 16; ++j) u.v[j] = (__bf16)p[(size_t)j * ld];
    uint4* q = (uint4*)(dst + (size_t)tile * TILE_ELEMS) + lane * 2;
    q[0] = u.q[0];
    q[1] = u.q[1];
}

// ---- init ------------------------------------------------------------------

__global__ void moe_init_kernel(float* __restrict__ out, int n_out_zero,
                                float* __restrict__ stats) {
    int i = blockIdx.x * 256 + threadIdx.x;
    if (i < n_out_zero) out[i] = 0.0f;
    if (i < 2) stats[i] = 0.0f;
}

// ---- gating: one wave32 per token ------------------------------------------

__device__ __forceinline__ float softplus_f(float v) {
    return (v > 20.0f) ? v : log1pf(expf(v));
}

__global__ __launch_bounds__(256)
void moe_gate_kernel(const float* __restrict__ x,
                     const float* __restrict__ noise,
                     const float* __restrict__ w_gate,
                     const float* __restrict__ w_noise,
                     float* __restrict__ gates,     // [B][E]
                     float* __restrict__ stats) {   // {sum_p, sum_p2}
    const int lane = threadIdx.x & 31;
    const int wave = threadIdx.x >> 5;
    const int b    = blockIdx.x * 8 + wave;
    if (b >= BB) return;

    float cg[EE], cn[EE];
#pragma unroll
    for (int e = 0; e < EE; ++e) { cg[e] = 0.0f; cn[e] = 0.0f; }

    const float* xr = x + (size_t)b * DD;
    for (int i = 0; i < DD / 32; ++i) {
        const int d = i * 32 + lane;
        const float xv = xr[d];
        const float* wg = w_gate  + (size_t)d * EE;
        const float* wn = w_noise + (size_t)d * EE;
#pragma unroll
        for (int e = 0; e < EE; ++e) {
            cg[e] += xv * wg[e];
            cn[e] += xv * wn[e];
        }
    }
#pragma unroll
    for (int off = 16; off > 0; off >>= 1) {
#pragma unroll
        for (int e = 0; e < EE; ++e) {
            cg[e] += __shfl_xor(cg[e], off, 32);
            cn[e] += __shfl_xor(cn[e], off, 32);
        }
    }

    if (lane == 0) {
        float clean[EE], sd[EE], noisy[EE], tmp[EE];
#pragma unroll
        for (int e = 0; e < EE; ++e) {
            clean[e] = cg[e];
            sd[e]    = softplus_f(cn[e]) + NOISE_EPS;
            noisy[e] = clean[e] + noise[(size_t)b * EE + e] * sd[e];
            tmp[e]   = noisy[e];
        }
        float tv[KK + 1]; int ti[KK + 1];
#pragma unroll
        for (int m = 0; m < KK + 1; ++m) {
            int best = 0; float bv = -3.4e38f;
#pragma unroll
            for (int e = 0; e < EE; ++e)
                if (tmp[e] > bv) { bv = tmp[e]; best = e; }
            tv[m] = bv; ti[m] = best; tmp[best] = -3.4e38f;
        }
        float s = 0.0f, ex[KK];
#pragma unroll
        for (int m = 0; m < KK; ++m) { ex[m] = expf(tv[m] - tv[0]); s += ex[m]; }
        float g[EE];
#pragma unroll
        for (int e = 0; e < EE; ++e) g[e] = 0.0f;
#pragma unroll
        for (int m = 0; m < KK; ++m) g[ti[m]] = ex[m] / s;
#pragma unroll
        for (int e = 0; e < EE; ++e) gates[(size_t)b * EE + e] = g[e];

        const float thr_in  = tv[KK];
        const float thr_out = tv[KK - 1];
        float s1 = 0.0f, s2 = 0.0f;
#pragma unroll
        for (int e = 0; e < EE; ++e) {
            const float thr = (noisy[e] > thr_in) ? thr_in : thr_out;
            const float z = (clean[e] - thr) / sd[e];
            const float p = 0.5f * (1.0f + erff(z * 0.70710678118654752f));
            s1 += p; s2 += p * p;
        }
        atomicAdd(&stats[0], s1);
        atomicAdd(&stats[1], s2);
    }
}

// ---- loss finalize ---------------------------------------------------------

__global__ void moe_loss_kernel(const float* __restrict__ stats,
                                float* __restrict__ loss_out) {
    const float N = (float)(BB * EE);
    const float mean = stats[0] / N;
    const float var  = (stats[1] - N * mean * mean) / (N - 1.0f);
    *loss_out = LOSS_COEF * var / (mean * mean + 1e-10f);
}

// ---- partial reduce: out = sum of expert-split partials --------------------

__global__ void moe_reduce_kernel(const float* __restrict__ part, int nparts,
                                  const int* __restrict__ expand_ptr,
                                  float* __restrict__ out) {
    const size_t i = (size_t)blockIdx.x * 256 + threadIdx.x;
    if (i >= (size_t)BB * OO) return;
    float s = 0.0f;
    for (int p = 0; p < nparts; ++p) s += part[(size_t)p * BB * OO + i];
    const int expand = expand_ptr[0];
    const int row = (int)(i >> 10);
    const int col = (int)(i & (OO - 1));
    out[(size_t)row * expand * OO + col] = s;
}

// ---- expert FFN: 32-token tile, 8 waves, gate folded into h ----------------
// grid.x = B/MT token tiles, grid.y = expert splits. Each block computes
// y_partial = sum_{e in split} g_e*(relu(x W1_e + b1_e) W2_e) + g_e b2_e
// and writes it to part[blockIdx.y].

template <bool SWZ>
__global__ __launch_bounds__(256)
void moe_ffn_kernel(const float* __restrict__ x,
                    const float* __restrict__ W1,
                    const float* __restrict__ b1,
                    const float* __restrict__ W2,
                    const float* __restrict__ b2,
                    const __bf16* __restrict__ swzW1,
                    const __bf16* __restrict__ swzW2,
                    const float* __restrict__ gates,
                    float* __restrict__ part,
                    int e_count) {
    __shared__ __bf16 xs[MT * DD];     // 64 KB: x tile (bf16)
    __shared__ __bf16 hbuf[MT * HC];   // 16 KB: hidden chunk (bf16)
    __shared__ float  gsh[MT * EE];    // 1 KB: gates for 32 tokens

    const int tid   = threadIdx.x;
    const int lane  = tid & 31;
    const int wave  = tid >> 5;
    const int tok0  = blockIdx.x * MT;
    const int col16 = lane & 15;
    const int hb8   = (lane < 16) ? 0 : 8;
    const int e_begin = blockIdx.y * e_count;
    float* dest = part + (size_t)blockIdx.y * BB * OO;

    for (int i = tid; i < MT * DD; i += 256) {
        const int r = i >> 10, c = i & (DD - 1);
        xs[i] = (__bf16)x[(size_t)(tok0 + r) * DD + c];
    }
    if (tid < MT * EE) gsh[tid] = gates[(size_t)tok0 * EE + tid];
    __syncthreads();

    f32x8 yacc[2][8];
#pragma unroll
    for (int rb = 0; rb < 2; ++rb)
#pragma unroll
        for (int nt = 0; nt < 8; ++nt) yacc[rb][nt] = (f32x8)0.0f;

    for (int ei = 0; ei < e_count; ++ei) {
        const int e = e_begin + ei;
        const float*  W1e = W1 + (size_t)e * DD * HH;
        const float*  W2e = W2 + (size_t)e * HH * OO;
        const __bf16* S1e = swzW1 + (size_t)e * KT1 * NT1 * TILE_ELEMS;
        const __bf16* S2e = swzW2 + (size_t)e * KT2 * NT2 * TILE_ELEMS;

        for (int hc = 0; hc < HH / HC; ++hc) {
            const int Hc = hc * HC;
            __syncthreads();  // hbuf WAR vs previous chunk's reads

            // -- phase 1: h[:, Hc + wave*32 .. +32] = g*relu(x @ W1 + b1) --
            {
                f32x8 acc[2][2];
#pragma unroll
                for (int rb = 0; rb < 2; ++rb)
#pragma unroll
                    for (int np = 0; np < 2; ++np) acc[rb][np] = (f32x8)0.0f;

                const int base_nt = (Hc + wave * 32) >> 4;
#pragma unroll 1
                for (int k0 = 0; k0 < DD; k0 += 32) {
                    const bf16x16 a0 = load_a_lds(xs, DD, k0, lane);
                    const bf16x16 a1 = load_a_lds(xs + 16 * DD, DD, k0, lane);
#pragma unroll
                    for (int np = 0; np < 2; ++np) {
                        bf16x16 bm;
                        if (SWZ) {
                            const __bf16* t = S1e +
                                (size_t)((k0 >> 5) * NT1 + base_nt + np) *
                                    TILE_ELEMS;
                            bm = load_b_swz(t, lane);
                        } else {
                            bm = load_b_f32(
                                W1e + (size_t)k0 * HH + Hc + wave * 32 + np * 16,
                                HH, lane);
                        }
                        acc[0][np] = wmma_bf16(a0, bm, acc[0][np]);
                        acc[1][np] = wmma_bf16(a1, bm, acc[1][np]);
                    }
                }
                // epilogue: bias, relu, fold gate, store bf16 h
#pragma unroll
                for (int rb = 0; rb < 2; ++rb)
#pragma unroll
                    for (int np = 0; np < 2; ++np) {
                        const int lcol = wave * 32 + np * 16;
                        const float bias =
                            b1[(size_t)e * HH + Hc + lcol + col16];
#pragma unroll
                        for (int v = 0; v < 8; ++v) {
                            const int row = rb * 16 + v + hb8;
                            float t = acc[rb][np][v] + bias;
                            t = t > 0.0f ? t : 0.0f;
                            t *= gsh[row * EE + e];
                            hbuf[row * HC + lcol + col16] = (__bf16)t;
                        }
                    }
            }
            __syncthreads();

            // -- phase 2: yacc += h_chunk @ W2[Hc:Hc+HC, wave cols] --------
#pragma unroll 1
            for (int k0 = 0; k0 < HC; k0 += 32) {
                const bf16x16 a0 = load_a_lds(hbuf, HC, k0, lane);
                const bf16x16 a1 = load_a_lds(hbuf + 16 * HC, HC, k0, lane);
#pragma unroll
                for (int nt = 0; nt < 8; ++nt) {
                    const int ocol = wave * 128 + nt * 16;
                    bf16x16 bm;
                    if (SWZ) {
                        const __bf16* t = S2e +
                            (size_t)(((Hc + k0) >> 5) * NT2 + (ocol >> 4)) *
                                TILE_ELEMS;
                        bm = load_b_swz(t, lane);
                    } else {
                        bm = load_b_f32(W2e + (size_t)(Hc + k0) * OO + ocol,
                                        OO, lane);
                    }
                    yacc[0][nt] = wmma_bf16(a0, bm, yacc[0][nt]);
                    yacc[1][nt] = wmma_bf16(a1, bm, yacc[1][nt]);
                }
            }
        }
    }

    // add sum_e g_e * b2_e (rank-1), then store partial
#pragma unroll
    for (int nt = 0; nt < 8; ++nt) {
        const int ocol = wave * 128 + nt * 16 + col16;
        for (int ei = 0; ei < e_count; ++ei) {
            const int e = e_begin + ei;
            const float b2v = b2[(size_t)e * OO + ocol];
#pragma unroll
            for (int rb = 0; rb < 2; ++rb)
#pragma unroll
                for (int v = 0; v < 8; ++v)
                    yacc[rb][nt][v] += gsh[(rb * 16 + v + hb8) * EE + e] * b2v;
        }
#pragma unroll
        for (int rb = 0; rb < 2; ++rb)
#pragma unroll
            for (int v = 0; v < 8; ++v) {
                const int row = tok0 + rb * 16 + v + hb8;
                dest[(size_t)row * OO + ocol] = yacc[rb][nt][v];
            }
    }
}

// ---- launcher --------------------------------------------------------------

extern "C" void kernel_launch(void* const* d_in, const int* in_sizes, int n_in,
                              void* d_out, int out_size, void* d_ws, size_t ws_size,
                              hipStream_t stream) {
    const float* x       = (const float*)d_in[0];
    const float* noise   = (const float*)d_in[1];
    const int*   expand  = (const int*)  d_in[2];
    const float* w_gate  = (const float*)d_in[3];
    const float* w_noise = (const float*)d_in[4];
    const float* W1      = (const float*)d_in[5];
    const float* b1      = (const float*)d_in[6];
    const float* W2      = (const float*)d_in[7];
    const float* b2      = (const float*)d_in[8];
    float* out = (float*)d_out;

    // ws layout: gates | stats | part[2] | swzW1 | swzW2
    float* gates_ws = (float*)d_ws;                       // B*E floats
    float* stats    = gates_ws + (size_t)BB * EE;
    const size_t part_off = ((size_t)BB * EE * 4 + 1024 + 255) & ~(size_t)255;
    float* part = (float*)((char*)d_ws + part_off);
    const size_t part_bytes = (size_t)2 * BB * OO * sizeof(float);
    const size_t swz_off = (part_off + part_bytes + 255) & ~(size_t)255;
    __bf16* swzW1 = (__bf16*)((char*)d_ws + swz_off);
    const size_t n_w1 = (size_t)EE * DD * HH;
    const size_t n_w2 = (size_t)EE * HH * OO;
    __bf16* swzW2 = swzW1 + n_w1;
    const bool use_swz =
        (ws_size >= swz_off + (n_w1 + n_w2) * sizeof(__bf16));

    const int n_out_zero = (out_size != BB * OO + 1) ? out_size : 0;
    const int init_elems = n_out_zero > 2 ? n_out_zero : 2;
    moe_init_kernel<<<(init_elems + 255) / 256, 256, 0, stream>>>(
        out, n_out_zero, stats);

    moe_gate_kernel<<<BB / 8, 256, 0, stream>>>(
        x, noise, w_gate, w_noise, gates_ws, stats);

    moe_loss_kernel<<<1, 1, 0, stream>>>(stats, out + (out_size - 1));

    int nparts;
    if (use_swz) {
        const int t1 = EE * KT1 * NT1;
        const int t2 = EE * KT2 * NT2;
        moe_swz_kernel<<<(t1 + 7) / 8, 256, 0, stream>>>(
            W1, swzW1, KT1, NT1, HH, t1);
        moe_swz_kernel<<<(t2 + 7) / 8, 256, 0, stream>>>(
            W2, swzW2, KT2, NT2, OO, t2);
        nparts = 2;  // 2-way expert split: 256 workgroups
        moe_ffn_kernel<true><<<dim3(BB / MT, nparts), 256, 0, stream>>>(
            x, W1, b1, W2, b2, swzW1, swzW2, gates_ws, part, EE / nparts);
    } else {
        nparts = 1;
        moe_ffn_kernel<false><<<dim3(BB / MT, nparts), 256, 0, stream>>>(
            x, W1, b1, W2, b2, swzW1, swzW2, gates_ws, part, EE);
    }

    moe_reduce_kernel<<<(BB * OO + 255) / 256, 256, 0, stream>>>(
        part, nparts, expand, out);
}
